// ResidualIndRNNBlock_77051713290732
// MI455X (gfx1250) — matrix operand order repaired and checked by reading, not compiled, not run on hardware
//
#include <hip/hip_runtime.h>
#include <hip/hip_bf16.h>

typedef __attribute__((ext_vector_type(16))) __bf16 bf16x16;
typedef __attribute__((ext_vector_type(8)))  __bf16 bf16x8;
typedef __attribute__((ext_vector_type(8)))  float  f32x8;
typedef __attribute__((ext_vector_type(4)))  unsigned int u32x4;
typedef __attribute__((ext_vector_type(4)))  int v4i;

#define LN_EPS 1e-5f
#define BDIM 32
#define TDIM 2048
#define DDIM 256
#define HDIM 512
#define NWAVES 8   // 256 threads / wave32

#if __has_builtin(__builtin_amdgcn_global_load_async_to_lds_b128) && \
    __has_builtin(__builtin_amdgcn_s_wait_asynccnt)
#define HAVE_ASYNC_LDS 1
#else
#define HAVE_ASYNC_LDS 0
#endif

// ---------------------------------------------------------------------------
// fp32 -> bf16 elementwise convert (grid-stride)
// ---------------------------------------------------------------------------
__global__ void k_cvt_bf16(const float* __restrict__ in, __bf16* __restrict__ out,
                           size_t n) {
    size_t i = (size_t)blockIdx.x * blockDim.x + threadIdx.x;
    size_t stride = (size_t)gridDim.x * blockDim.x;
    for (; i < n; i += stride) out[i] = (__bf16)in[i];
}

// fp32 [R,C] -> bf16 [C,R] transpose (small matrices only)
__global__ void k_transpose_bf16(const float* __restrict__ in, __bf16* __restrict__ out,
                                 int R, int C) {
    int i = blockIdx.x * blockDim.x + threadIdx.x;
    if (i < R * C) {
        int r = i / C, c = i % C;
        out[(size_t)c * R + r] = (__bf16)in[i];
    }
}

// ---------------------------------------------------------------------------
// bf16 GEMM, f32 accumulate:  C[M,N] = A[M,K] * Bt[N,K]^T (+ bias[N])
// Block: 256 threads = 8 waves; block tile 128(M) x 64(N); wave tile 16x64.
// B panel [64][K] staged in LDS once per block (async global->LDS, ASYNCcnt),
// row stride padded 16B to spread 16-lane row-parallel ds_load_b128 across
// banks. A fragment software-pipelined one k-step ahead; all four B
// fragments of a k-step issued before the WMMA batch so LDS latency is paid
// once per k-step, not per WMMA.
// ---------------------------------------------------------------------------
template <int K>
__global__ __launch_bounds__(256)
void k_gemm_bf16(const __bf16* __restrict__ A,    // [M,K]
                 const __bf16* __restrict__ Bt,   // [N,K]
                 const float*  __restrict__ bias, // [N] or null
                 float*        __restrict__ Cf,   // [M,N] f32
                 __bf16*       __restrict__ Cbf,  // [M,N] bf16 or null
                 int N) {
    constexpr int LDB = K + 8;                    // +16B pad per row
    __shared__ __bf16 sB[64 * LDB];

    const int tid  = threadIdx.x;
    const int lane = tid & 31;
    const int wave = tid >> 5;
    const int l16  = lane & 15;
    const int hi   = (lane >> 4) & 1;
    const int n0   = blockIdx.x * 64;
    const int mw   = blockIdx.y * 128 + wave * 16;

    // ---- stage B panel [64][K] -> LDS (16B chunks, all 256 threads)
    constexpr int CHUNKS = 64 * K / 8;            // 16-byte chunks
#pragma unroll
    for (int c = tid; c < CHUNKS; c += 256) {
        const int row = c / (K / 8);
        const int col = (c % (K / 8)) * 8;
        const __bf16* g = Bt + (size_t)(n0 + row) * K + col;
        __bf16* s = sB + row * LDB + col;
#if HAVE_ASYNC_LDS
        __builtin_amdgcn_global_load_async_to_lds_b128(
            (v4i*)(void*)g, (v4i*)(void*)s, /*offset=*/0, /*cpol=*/0);
#else
        *(u32x4*)s = *(const u32x4*)g;
#endif
    }
#if HAVE_ASYNC_LDS
    __builtin_amdgcn_s_wait_asynccnt(0);
#endif
    __syncthreads();

    // ---- main K loop
    const __bf16* arow = A + (size_t)(mw + l16) * K;
    const int kb = hi * 8;

    f32x8 acc[4] = {};

    // A pipeline: preload k-step 0
    bf16x8 alo = *(const bf16x8*)(arow + kb);
    bf16x8 ahi = *(const bf16x8*)(arow + kb + 16);

#pragma unroll 2
    for (int k0 = 0; k0 < K; k0 += 32) {
        bf16x16 a;
#pragma unroll
        for (int e = 0; e < 8; ++e) { a[e] = alo[e]; a[e + 8] = ahi[e]; }

        // issue ALL B fragment loads for this k-step up front (8x ds_load_b128)
        bf16x16 bfr[4];
#pragma unroll
        for (int j = 0; j < 4; ++j) {
            const __bf16* bl = sB + (j * 16 + l16) * LDB + k0 + hi * 8;
            bf16x8 blo = *(const bf16x8*)(bl);
            bf16x8 bhi = *(const bf16x8*)(bl + 16);
#pragma unroll
            for (int e = 0; e < 8; ++e) { bfr[j][e] = blo[e]; bfr[j][e + 8] = bhi[e]; }
        }

        if (k0 + 32 < K) {      // prefetch next k-step's A before the WMMAs
            alo = *(const bf16x8*)(arow + k0 + 32 + kb);
            ahi = *(const bf16x8*)(arow + k0 + 32 + kb + 16);
        }

#pragma unroll
        for (int j = 0; j < 4; ++j) {
            acc[j] = __builtin_amdgcn_wmma_f32_16x16x32_bf16(
                /*neg_a=*/false, a, /*neg_b=*/false, bfr[j],
                /*c_mod=*/(short)0, acc[j],
                /*reuse_a=*/false, /*reuse_b=*/false);
        }
    }

    // ---- epilogue: lane holds col n=lane%16, rows mbase+r, mbase=mw+hi*8
#pragma unroll
    for (int j = 0; j < 4; ++j) {
        const int n  = n0 + j * 16 + l16;
        const float bv = bias ? bias[n] : 0.0f;
        const int mbase = mw + hi * 8;
#pragma unroll
        for (int r = 0; r < 8; ++r) {
            float v = acc[j][r] + bv;
            size_t idx = (size_t)(mbase + r) * N + n;
            Cf[idx] = v;
            if (Cbf) Cbf[idx] = (__bf16)v;
        }
    }
}

// ---------------------------------------------------------------------------
// Sequential IndRNN recurrence. One block per batch row b.
// 256 threads, 2 H-elements per thread. h state carried in registers.
// LayerNorm via single-pass (sum, sumsq) wave shfl reduction + LDS combine.
// `ic` aliases `out`: each thread reads its ic elements before overwriting.
// ---------------------------------------------------------------------------
__device__ inline void block_reduce2(float& s, float& q, float* sS, float* sQ) {
#pragma unroll
    for (int off = 16; off > 0; off >>= 1) {
        s += __shfl_xor(s, off, 32);
        q += __shfl_xor(q, off, 32);
    }
    const int wave = threadIdx.x >> 5;
    const int lane = threadIdx.x & 31;
    __syncthreads();                       // protect LDS reuse across calls
    if (lane == 0) { sS[wave] = s; sQ[wave] = q; }
    __syncthreads();
    s = 0.0f; q = 0.0f;
#pragma unroll
    for (int w = 0; w < NWAVES; ++w) { s += sS[w]; q += sQ[w]; }
}

__global__ __launch_bounds__(256)
void k_recurrence(const float* __restrict__ xp,   // [B,T,H]
                  const float* ic,                // [B,T,H] (aliases out)
                  const float* __restrict__ wr,   // [H]
                  const float* __restrict__ bs,   // [H]
                  const float* __restrict__ g1v, const float* __restrict__ b1v,
                  const float* __restrict__ g2v, const float* __restrict__ b2v,
                  float* out) {
    __shared__ float sS[NWAVES], sQ[NWAVES];
    const int b  = blockIdx.x;
    const int h0 = threadIdx.x * 2;
    const float invH = 1.0f / (float)HDIM;

    float w[2], bi[2], ga1[2], be1[2], ga2[2], be2[2];
#pragma unroll
    for (int i = 0; i < 2; ++i) {
        w[i]   = wr[h0 + i];
        bi[i]  = bs[h0 + i];
        ga1[i] = g1v[h0 + i];
        be1[i] = b1v[h0 + i];
        ga2[i] = g2v[h0 + i];
        be2[i] = b2v[h0 + i];
    }

    float h[2] = {0.0f, 0.0f};

    for (int t = 0; t < TDIM; ++t) {
        const size_t base = ((size_t)b * TDIM + t) * HDIM + h0;
        float icv[2] = {ic[base], ic[base + 1]};
        float xpv[2] = {xp[base], xp[base + 1]};
        if (t + 1 < TDIM) {                 // emits global_prefetch_b8
            __builtin_prefetch(ic + base + HDIM, 0, 3);
            __builtin_prefetch(xp + base + HDIM, 0, 3);
        }

        // LN1 input: v = ic + h*w + bias
        float v[2], s = 0.0f, q = 0.0f;
#pragma unroll
        for (int i = 0; i < 2; ++i) {
            v[i] = icv[i] + h[i] * w[i] + bi[i];
            s += v[i]; q += v[i] * v[i];
        }
        block_reduce2(s, q, sS, sQ);
        float mean = s * invH;
        float var  = fmaxf(q * invH - mean * mean, 0.0f);
        float rs   = rsqrtf(var + LN_EPS);

        // h1 = relu(LN1(v)); LN2 input: u = h1*w + ic
        float u[2]; s = 0.0f; q = 0.0f;
#pragma unroll
        for (int i = 0; i < 2; ++i) {
            float h1 = fmaxf((v[i] - mean) * rs * ga1[i] + be1[i], 0.0f);
            u[i] = h1 * w[i] + icv[i];
            s += u[i]; q += u[i] * u[i];
        }
        block_reduce2(s, q, sS, sQ);
        mean = s * invH;
        var  = fmaxf(q * invH - mean * mean, 0.0f);
        rs   = rsqrtf(var + LN_EPS);

#pragma unroll
        for (int i = 0; i < 2; ++i) {
            float h2 = fmaxf((u[i] - mean) * rs * ga2[i] + be2[i], 0.0f);
            float hn = h2 + xpv[i];
            out[base + i] = hn;
            h[i] = hn;
        }
    }
}

// ---------------------------------------------------------------------------
extern "C" void kernel_launch(void* const* d_in, const int* in_sizes, int n_in,
                              void* d_out, int out_size, void* d_ws, size_t ws_size,
                              hipStream_t stream) {
    const float* x       = (const float*)d_in[0];  // [B,T,D]
    const float* W_proj  = (const float*)d_in[1];  // [H,D]
    const float* b_proj  = (const float*)d_in[2];  // [H]
    const float* W_in    = (const float*)d_in[3];  // [H,H]
    const float* wr      = (const float*)d_in[4];  // [H]
    const float* bias    = (const float*)d_in[5];  // [H]
    const float* gamma1  = (const float*)d_in[6];
    const float* beta1   = (const float*)d_in[7];
    const float* gamma2  = (const float*)d_in[8];
    const float* beta2   = (const float*)d_in[9];
    float* out = (float*)d_out;                    // [B,T,H]

    const int BT = BDIM * TDIM;                    // 65536 rows

    // workspace layout
    char* ws = (char*)d_ws;
    size_t off = 0;
    __bf16* A1  = (__bf16*)(ws + off); off += (size_t)BT * DDIM * sizeof(__bf16);   // x bf16
    __bf16* Bt1 = (__bf16*)(ws + off); off += (size_t)HDIM * DDIM * sizeof(__bf16); // W_proj [n][k]
    __bf16* Bt2 = (__bf16*)(ws + off); off += (size_t)HDIM * HDIM * sizeof(__bf16); // W_in^T [g][h]
    float*  XPf = (float*) (ws + off); off += (size_t)BT * HDIM * sizeof(float);    // xp f32
    __bf16* XPb = (__bf16*)(ws + off); off += (size_t)BT * HDIM * sizeof(__bf16);   // xp bf16

    // 1) precision conversions / small transpose
    k_cvt_bf16<<<4096, 256, 0, stream>>>(x, A1, (size_t)BT * DDIM);
    // W_proj is [H,D] row-major == desired [N=H][K=D] layout: plain convert
    k_cvt_bf16<<<512, 256, 0, stream>>>(W_proj, Bt1, (size_t)HDIM * DDIM);
    // input_weights is [h][g]; need Bt2[g][h]: transpose
    k_transpose_bf16<<<(HDIM * HDIM + 255) / 256, 256, 0, stream>>>(W_in, Bt2, HDIM, HDIM);

    // 2) GEMM1: xp = x @ W_proj^T + b_proj  (M=65536, N=512, K=256)
    dim3 grid1(HDIM / 64, BT / 128);
    k_gemm_bf16<DDIM><<<grid1, 256, 0, stream>>>(A1, Bt1, b_proj, XPf, XPb, HDIM);

    // 3) GEMM2: ic = xp @ input_weights     (M=65536, N=512, K=512) -> into d_out
    k_gemm_bf16<HDIM><<<grid1, 256, 0, stream>>>(XPb, Bt2, nullptr, out, nullptr, HDIM);

    // 4) sequential recurrence: one block per batch row, ic read from / out
    //    written to the same buffer (read-before-write per thread per step)
    k_recurrence<<<BDIM, 256, 0, stream>>>(XPf, out, wr, bias,
                                           gamma1, beta1, gamma2, beta2, out);
}
